// StructuredGlobalConvolution_40295383171695
// MI455X (gfx1250) — matrix operand order repaired and checked by reading, not compiled, not run on hardware
//
#include <hip/hip_runtime.h>
#include <hip/hip_fp16.h>
#include <math.h>

#define BATCH      8
#define CHANNELS   256
#define LENGTH     16384
#define KSIZE      64
#define NUM_SCALES 9
#define NFFT       32768
#define LOGN       15
#define FTH        1024
#define PI_F       3.14159265358979323846f

typedef __attribute__((ext_vector_type(16))) _Float16 v16h;
typedef __attribute__((ext_vector_type(8)))  _Float16 v8h;
typedef __attribute__((ext_vector_type(8)))  float    v8f;

__device__ __forceinline__ float gelu_exact(float x) {
    return 0.5f * x * (1.0f + erff(x * 0.70710678118654752f));
}

// ---------------------------------------------------------------------------
// Kernel A: build normalized multi-scale kernel k_full[C][L].
// ---------------------------------------------------------------------------
__global__ void build_k(const float* __restrict__ kin, float* __restrict__ kfull) {
    const int c   = blockIdx.x;
    const int tid = threadIdx.x;
    const float mult = 1.0f + 3.0f * (float)c / (float)(CHANNELS - 1);

    __shared__ float red[256];
    float ss = 0.0f;

    for (int t = tid; t < LENGTH; t += 256) {
        int s, off, outlen;
        if (t < 64) { s = 0; off = t; outlen = 64; }
        else {
            int hb = 31 - __clz(t);      // floor(log2(t)), t in [64, 16384)
            s = hb - 5;                  // segment index 1..8
            outlen = 64 << (s - 1);      // == segment start
            off = t - outlen;
        }
        const float* src = kin + ((size_t)s * CHANNELS + c) * KSIZE;
        float val;
        if (outlen == 64) {
            val = src[off];
        } else {
            float scale = 64.0f / (float)outlen;
            float coord = ((float)off + 0.5f) * scale - 0.5f;
            coord = fminf(fmaxf(coord, 0.0f), 63.0f);
            int lo = (int)floorf(coord);
            int hi = min(lo + 1, 63);
            float w = coord - (float)lo;
            val = src[lo] * (1.0f - w) + src[hi] * w;
        }
        val *= powf(mult, (float)(NUM_SCALES - 1 - s));
        kfull[(size_t)c * LENGTH + t] = val;
        ss += val * val;
    }

    red[tid] = ss;
    __syncthreads();
    for (int s2 = 128; s2 > 0; s2 >>= 1) {
        if (tid < s2) red[tid] += red[tid + s2];
        __syncthreads();
    }
    const float rn = 1.0f / sqrtf(red[0]);
    for (int t = tid; t < LENGTH; t += 256) kfull[(size_t)c * LENGTH + t] *= rn;
}

// ---------------------------------------------------------------------------
// Kernel B: W (f32, row-major [v][u]) -> f16
// ---------------------------------------------------------------------------
__global__ void conv_w_f16(const float* __restrict__ W, _Float16* __restrict__ Wh) {
    int i = blockIdx.x * blockDim.x + threadIdx.x;
    if (i < CHANNELS * CHANNELS) Wh[i] = (_Float16)W[i];
}

// ---------------------------------------------------------------------------
// Kernel C: FFT convolution per (b,c), two-for-one trick z = u + i*k.
// Initial tile load uses CDNA5 GLOBAL_LOAD_ASYNC_TO_LDS_B128 (ASYNCcnt).
// ---------------------------------------------------------------------------
__global__ void fftconv(const float* __restrict__ u, const float* __restrict__ kfull,
                        const float* __restrict__ D, _Float16* __restrict__ gt) {
    extern __shared__ float sm[];
    float* re = sm;
    float* im = sm + NFFT;

    const int bc  = blockIdx.x;
    const int b   = bc / CHANNELS;
    const int c   = bc % CHANNELS;
    const int tid = threadIdx.x;

    const float* up = u + ((size_t)b * CHANNELS + c) * LENGTH;
    const float* kp = kfull + (size_t)c * LENGTH;

    // ---- async global->LDS load of u and k (each lane moves 16B per op) ----
    {
        const unsigned dynbase = __builtin_amdgcn_groupstaticsize(); // dynamic LDS start
#pragma unroll
        for (int it = 0; it < LENGTH / (FTH * 4); ++it) {
            const int i = (tid + it * FTH) * 4;                 // float index
            const unsigned lre = dynbase + (unsigned)(i * 4);            // &re[i]
            const unsigned lim = dynbase + (unsigned)((NFFT + i) * 4);   // &im[i]
            asm volatile("global_load_async_to_lds_b128 %0, %1, off"
                         :: "v"(lre), "v"(up + i) : "memory");
            asm volatile("global_load_async_to_lds_b128 %0, %1, off"
                         :: "v"(lim), "v"(kp + i) : "memory");
        }
        // zero-pad upper halves with regular LDS stores
        for (int i = LENGTH + tid; i < NFFT; i += FTH) { re[i] = 0.0f; im[i] = 0.0f; }
        asm volatile("s_wait_asynccnt 0x0" ::: "memory");
    }
    __syncthreads();

    // ---- forward DIF: natural input -> bit-reversed spectrum ----
    for (int stage = 0; stage < LOGN; ++stage) {
        const int shift = (LOGN - 1) - stage;      // half = 1 << shift
        const int half  = 1 << shift;
        const float astep = -PI_F / (float)half;

        if (half <= FTH) {
            // off constant across this thread's butterflies: one exact sincos
            const int off = tid & (half - 1);
            float tw_i, tw_r;
            __sincosf(astep * (float)off, &tw_i, &tw_r);
#pragma unroll
            for (int m = 0; m < NFFT / 2 / FTH; ++m) {
                const int j  = tid + (m << 10);
                const int i0 = ((j >> shift) << (shift + 1)) + off;
                const int i1 = i0 + half;
                float ar = re[i0], ai = im[i0];
                float br = re[i1], bi = im[i1];
                float dr = ar - br, di = ai - bi;
                re[i0] = ar + br; im[i0] = ai + bi;
                re[i1] = dr * tw_r - di * tw_i;
                im[i1] = dr * tw_i + di * tw_r;
            }
        } else {
            // recurrence: w = exp(i*astep*j); true twiddle = (-1)^(j>>shift) * w
            float w_r, w_i, s_r, s_i;
            __sincosf(astep * (float)tid, &w_i, &w_r);
            __sincosf(astep * (float)FTH, &s_i, &s_r);
#pragma unroll
            for (int m = 0; m < NFFT / 2 / FTH; ++m) {
                const int j   = tid + (m << 10);
                const int blk = j >> shift;
                const int off = j & (half - 1);
                const int i0  = (blk << (shift + 1)) + off;
                const int i1  = i0 + half;
                float tw_r = w_r, tw_i = w_i;
                if (blk & 1) { tw_r = -tw_r; tw_i = -tw_i; }
                float ar = re[i0], ai = im[i0];
                float br = re[i1], bi = im[i1];
                float dr = ar - br, di = ai - bi;
                re[i0] = ar + br; im[i0] = ai + bi;
                re[i1] = dr * tw_r - di * tw_i;
                im[i1] = dr * tw_i + di * tw_r;
                float n_r = w_r * s_r - w_i * s_i;
                float n_i = w_r * s_i + w_i * s_r;
                w_r = n_r; w_i = n_i;
            }
        }
        __syncthreads();
    }

    // ---- disentangle U,K (Hermitian split) and multiply: H = U*K ----
    for (int f = tid; f <= NFFT / 2; f += FTH) {
        const int fr = (NFFT - f) & (NFFT - 1);
        const unsigned bf  = __brev((unsigned)f)  >> (32 - LOGN);
        const unsigned bfr = __brev((unsigned)fr) >> (32 - LOGN);
        float zfr = re[bf],  zfi = im[bf];
        float zrr = re[bfr], zri = im[bfr];
        float ur = 0.5f * (zfr + zrr), ui = 0.5f * (zfi - zri);
        float kr = 0.5f * (zfi + zri), ki = 0.5f * (zrr - zfr);
        float hr = ur * kr - ui * ki;
        float hi = ur * ki + ui * kr;   // == 0 at f==0, f==N/2 (self-paired)
        re[bf]  = hr;  im[bf]  = hi;
        re[bfr] = hr;  im[bfr] = -hi;
    }
    __syncthreads();

    // ---- inverse DIT: bit-reversed input -> natural output ----
    for (int stage = 0; stage < LOGN; ++stage) {
        const int half  = 1 << stage;
        const float astep = PI_F / (float)half;

        if (half <= FTH) {
            const int off = tid & (half - 1);
            float tw_i, tw_r;
            __sincosf(astep * (float)off, &tw_i, &tw_r);
#pragma unroll
            for (int m = 0; m < NFFT / 2 / FTH; ++m) {
                const int j  = tid + (m << 10);
                const int i0 = ((j >> stage) << (stage + 1)) + off;
                const int i1 = i0 + half;
                float br = re[i1], bi = im[i1];
                float tr = br * tw_r - bi * tw_i;
                float ti = br * tw_i + bi * tw_r;
                float ar = re[i0], ai = im[i0];
                re[i0] = ar + tr; im[i0] = ai + ti;
                re[i1] = ar - tr; im[i1] = ai - ti;
            }
        } else {
            float w_r, w_i, s_r, s_i;
            __sincosf(astep * (float)tid, &w_i, &w_r);
            __sincosf(astep * (float)FTH, &s_i, &s_r);
#pragma unroll
            for (int m = 0; m < NFFT / 2 / FTH; ++m) {
                const int j   = tid + (m << 10);
                const int blk = j >> stage;
                const int off = j & (half - 1);
                const int i0  = (blk << (stage + 1)) + off;
                const int i1  = i0 + half;
                float tw_r = w_r, tw_i = w_i;
                if (blk & 1) { tw_r = -tw_r; tw_i = -tw_i; }
                float br = re[i1], bi = im[i1];
                float tr = br * tw_r - bi * tw_i;
                float ti = br * tw_i + bi * tw_r;
                float ar = re[i0], ai = im[i0];
                re[i0] = ar + tr; im[i0] = ai + ti;
                re[i1] = ar - tr; im[i1] = ai - ti;
                float n_r = w_r * s_r - w_i * s_i;
                float n_i = w_r * s_i + w_i * s_r;
                w_r = n_r; w_i = n_i;
            }
        }
        __syncthreads();
    }

    // ---- y = conv/N + u*D ; gelu ; store f16 transposed [b][l][c].
    // Stride-C 2B scatter, but gt (64 MB) merges in the 192 MB L2.
    const float Dc   = D[c];
    const float invN = 1.0f / (float)NFFT;
    for (int l = tid; l < LENGTH; l += FTH) {
        float y = re[l] * invN + up[l] * Dc;
        gt[((size_t)b * LENGTH + l) * CHANNELS + c] = (_Float16)gelu_exact(y);
    }
}

// ---------------------------------------------------------------------------
// Kernel D: channel-mixing GEMM with WMMA f16->f32, software-pipelined.
// Out[b][v][l] = gelu( sum_u W[v][u]*g[b][u][l] + bias[v] )
// ---------------------------------------------------------------------------
__global__ void mixgemm(const _Float16* __restrict__ gt, const _Float16* __restrict__ Wh,
                        const float* __restrict__ bias, float* __restrict__ out) {
    const int blk   = blockIdx.x;
    const int b     = blk >> 10;                // LENGTH/16 = 1024 tiles per batch
    const int l0    = (blk & 1023) << 4;
    const int wave  = threadIdx.x >> 5;
    const int lane  = threadIdx.x & 31;
    const int mrow  = lane & 15;
    const int khalf = lane >> 4;

    // A fragment base: lanes 0-15 row m K=+0..15, lanes 16-31 K=+16..31
    const _Float16* apbase = gt + ((size_t)b * LENGTH + l0 + mrow) * CHANNELS + khalf * 16;
    // B fragment bases: col n = mrow, lanes 0-15 K=+0..15, lanes 16-31 K=+16..31
    const _Float16* bp[4];
#pragma unroll
    for (int t = 0; t < 4; ++t)
        bp[t] = Wh + (size_t)(((wave + (t << 2)) << 4) + mrow) * CHANNELS + khalf * 16;

    // CDNA5 prefetch of this tile's A row (emits global_prefetch_b8)
    __builtin_prefetch(apbase, 0, 0);
    __builtin_prefetch(apbase + 128, 0, 0);

    v8f acc[4] = {};

    // prologue: fragments for ks = 0
    v8h a_lo = *(const v8h*)(apbase);
    v8h a_hi = *(const v8h*)(apbase + 8);
    v16h a_cur = __builtin_shufflevector(a_lo, a_hi,
                    0,1,2,3,4,5,6,7,8,9,10,11,12,13,14,15);
    v16h b_cur[4];
#pragma unroll
    for (int t = 0; t < 4; ++t) b_cur[t] = *(const v16h*)(bp[t]);

#pragma unroll
    for (int kk = 0; kk < CHANNELS / 32; ++kk) {
        v16h a_nxt = a_cur;
        v16h b_nxt[4] = { b_cur[0], b_cur[1], b_cur[2], b_cur[3] };
        if (kk + 1 < CHANNELS / 32) {
            const int ks = (kk + 1) << 5;
            v8h lo = *(const v8h*)(apbase + ks);
            v8h hi = *(const v8h*)(apbase + ks + 8);
            a_nxt = __builtin_shufflevector(lo, hi,
                        0,1,2,3,4,5,6,7,8,9,10,11,12,13,14,15);
#pragma unroll
            for (int t = 0; t < 4; ++t) b_nxt[t] = *(const v16h*)(bp[t] + ks);
        }
#pragma unroll
        for (int t = 0; t < 4; ++t) {
            acc[t] = __builtin_amdgcn_wmma_f32_16x16x32_f16(
                false, a_cur, false, b_cur[t], (short)0, acc[t], false, false);
        }
        a_cur = a_nxt;
#pragma unroll
        for (int t = 0; t < 4; ++t) b_cur[t] = b_nxt[t];
    }

    // C/D layout: VGPR p -> M = p (lanes 0-15) or p+8 (lanes 16-31), N = mrow.
    // Per lane the 8 outputs are consecutive in l -> two b128 stores.
#pragma unroll
    for (int t = 0; t < 4; ++t) {
        const int v  = ((wave + (t << 2)) << 4) + mrow;
        const float bv = bias[v];
#pragma unroll
        for (int p = 0; p < 8; ++p) {
            const int m = p + (khalf << 3);
            float x = acc[t][p] + bv;
            out[((size_t)b * CHANNELS + v) * LENGTH + l0 + m] = gelu_exact(x);
        }
    }
}

// ---------------------------------------------------------------------------
extern "C" void kernel_launch(void* const* d_in, const int* in_sizes, int n_in,
                              void* d_out, int out_size, void* d_ws, size_t ws_size,
                              hipStream_t stream) {
    (void)in_sizes; (void)n_in; (void)out_size; (void)ws_size;

    const float* u   = (const float*)d_in[0];   // (8,256,16384)
    const float* kin = (const float*)d_in[1];   // (9,1,256,64)
    const float* D   = (const float*)d_in[2];   // (1,256)
    const float* W   = (const float*)d_in[3];   // (256,256)
    const float* bia = (const float*)d_in[4];   // (256,)
    float* out = (float*)d_out;                 // (8,256,16384) f32

    // workspace layout: gt f16 (64 MiB) | k_full f32 (16 MiB) | Wh f16 (128 KiB)
    char* ws = (char*)d_ws;
    const size_t gt_bytes = (size_t)BATCH * LENGTH * CHANNELS * sizeof(_Float16);
    const size_t kf_bytes = (size_t)CHANNELS * LENGTH * sizeof(float);
    _Float16* gt  = (_Float16*)ws;
    float*    kfl = (float*)(ws + gt_bytes);
    _Float16* Wh  = (_Float16*)(ws + gt_bytes + kf_bytes);

    build_k<<<CHANNELS, 256, 0, stream>>>(kin, kfl);
    conv_w_f16<<<(CHANNELS * CHANNELS + 255) / 256, 256, 0, stream>>>(W, Wh);
    fftconv<<<BATCH * CHANNELS, FTH, NFFT * 2 * sizeof(float), stream>>>(u, kfl, D, gt);
    mixgemm<<<BATCH * (LENGTH / 16), 128, 0, stream>>>(gt, Wh, bia, out);
}
